// GCN_16114717295067
// MI455X (gfx1250) — matrix operand order repaired and checked by reading, not compiled, not run on hardware
//
#include <hip/hip_runtime.h>

typedef __attribute__((ext_vector_type(16))) __bf16 v16bf;
typedef __attribute__((ext_vector_type(8)))  float  v8f;

#define EMB 64
#define HID 256
#define NT  16        // 256 / 16 column tiles
#define KTX 2         // 64  / 32 k-tiles (layer 0)
#define KTH 8         // 256 / 32 k-tiles (layers 1,2)
#define FRAG 512      // bf16 elems per 32x16 fragment tile (32 lanes x 16)
#define RT  4         // row tiles per block -> M = 64 rows/block
#define TT  2         // column tiles per wave (8 waves x 2 = 16 tiles)
#define NWAVE 8
#define BLK (NWAVE * 32)

// ---- bf16 hi/lo split: hi = truncation (exact residual), lo = RNE(residual).
// hi-store and lo-store both come straight from bits[31:16] (d16_hi stores).
__device__ __forceinline__ void split_bits(float v, unsigned& uhi, unsigned& ulo) {
  unsigned u = __float_as_uint(v);
  uhi = u & 0xFFFF0000u;                       // bf16 hi (truncated), exact fp32
  float res = v - __uint_as_float(uhi);        // exact residual
  unsigned r = __float_as_uint(res);
  ulo = r + 0x7FFFu + ((r >> 16) & 1u);        // RNE; bf16 lo in bits[31:16]
}

// ---------------- Kernel 0: weight prep ----------------
// Split fp32 W[K x N] into bf16 hi/lo, stored B-fragment-swizzled:
// out[((kt*NT + nt)*FRAG) + lane*16 + elem], with (ISA 7.12.2, 16-bit B):
//   n = nt*16 + lane%16 ;  k = kt*32 + (elem/8)*16 + (lane/16)*8 + elem%8
// -> each lane's fragment slice is 32 contiguous bytes (2x b128, coalesced).
__global__ __launch_bounds__(256) void gcn_wprep_kernel(
    const float* __restrict__ W, unsigned short* __restrict__ Wh,
    unsigned short* __restrict__ Wl, int K, int N) {
  int gid = blockIdx.x * blockDim.x + threadIdx.x;
  if (gid >= K * N) return;
  int elem = gid & 15;
  int lane = (gid >> 4) & 31;
  int tile = gid >> 9;
  int nt = tile & (NT - 1);
  int kt = tile >> 4;           // NT == 16
  int n = nt * 16 + (lane & 15);
  int k = kt * 32 + (elem >> 3) * 16 + (lane >> 4) * 8 + (elem & 7);
  float v = W[(size_t)k * N + n];
  unsigned uhi, ulo;
  split_bits(v, uhi, ulo);
  Wh[gid] = (unsigned short)(uhi >> 16);
  Wl[gid] = (unsigned short)(ulo >> 16);
}

// ---------------- Kernel 1: SpMM scatter-add ----------------
// x[row[e]] += val[e] * emb[col[e]]  (fp32 exact; emb/x are L2-resident)
__global__ __launch_bounds__(256) void gcn_spmm_kernel(
    const int* __restrict__ row, const int* __restrict__ col,
    const float* __restrict__ val, const float* __restrict__ emb,
    float* __restrict__ x, int E) {
  long gid = (long)blockIdx.x * blockDim.x + threadIdx.x;
  int e = (int)(gid >> 4);
  int j = (int)(gid & 15);
  if (e >= E) return;
  int r = row[e];
  int c = col[e];
  float v = val[e];
  const float4* erow = (const float4*)(emb + (size_t)c * EMB);
  float4 m = erow[j];
  float* xr = x + (size_t)r * EMB + j * 4;
  atomicAdd(xr + 0, v * m.x);
  atomicAdd(xr + 1, v * m.y);
  atomicAdd(xr + 2, v * m.z);
  atomicAdd(xr + 3, v * m.w);
}

// ---------------- Kernel 2: fused 3-layer MLP, bf16x3 split WMMA ----------------
// A-fragment LDS layout (ISA 7.12.2, 16-bit A 16x32):
//   lane L: M = L%16 ; elems 0..7  hold K = (L/16)*8 + 0..7,
//                      elems 8..15 hold K = 16 + (L/16)*8 + 0..7
__device__ __forceinline__ void store_act_frag(
    unsigned short* __restrict__ Dh, unsigned short* __restrict__ Dl,
    int m, int k, float v) {
  int kt = k >> 5, kin = k & 31;
  int l2 = m + (((kin >> 3) & 1) << 4);
  int e  = ((kin >> 4) << 3) + (kin & 7);
  int idx = kt * FRAG + l2 * 16 + e;
  unsigned uhi, ulo;
  split_bits(v, uhi, ulo);
  Dh[idx] = (unsigned short)(uhi >> 16);
  Dl[idx] = (unsigned short)(ulo >> 16);
}

// One layer over RT row tiles x TT column tiles per wave.
// A buffers: [rt][KT][lane][16] (hi/lo). Weights: fragment-swizzled global.
// D buffers always have KTH k-tiles (next layer K = 256).
template <int KT, bool RELU_TO_LDS>
__device__ __forceinline__ void gcn_layer(
    const unsigned short* __restrict__ Ah, const unsigned short* __restrict__ Al,
    const unsigned short* __restrict__ Wh, const unsigned short* __restrict__ Wl,
    unsigned short* __restrict__ Dh, unsigned short* __restrict__ Dl,
    float* __restrict__ outp, int r0, int Nrows, int lane, int wave) {
  const int mn = lane & 15;
  const int mrow = (lane >> 4) * 8;

  v8f acc[RT][TT] = {};
  for (int kt = 0; kt < KT; ++kt) {
    v16bf ah[RT], al[RT];
#pragma unroll
    for (int rt = 0; rt < RT; ++rt) {
      const int aoff = (rt * KT + kt) * FRAG + lane * 16;
      ah[rt] = *(const v16bf*)(Ah + aoff);
      al[rt] = *(const v16bf*)(Al + aoff);
    }
#pragma unroll
    for (int t = 0; t < TT; ++t) {
      const int nt = wave + t * NWAVE;
      const size_t boff = (size_t)(kt * NT + nt) * FRAG + lane * 16;
      v16bf bh = *(const v16bf*)(Wh + boff);
      v16bf bl = *(const v16bf*)(Wl + boff);
#pragma unroll
      for (int rt = 0; rt < RT; ++rt) {
        // bf16x3 split product: hi*hi + hi*lo + lo*hi (fp32 accumulate)
        acc[rt][t] = __builtin_amdgcn_wmma_f32_16x16x32_bf16(
            false, ah[rt], false, bh, (short)0, acc[rt][t], false, false);
        acc[rt][t] = __builtin_amdgcn_wmma_f32_16x16x32_bf16(
            false, ah[rt], false, bl, (short)0, acc[rt][t], false, false);
        acc[rt][t] = __builtin_amdgcn_wmma_f32_16x16x32_bf16(
            false, al[rt], false, bh, (short)0, acc[rt][t], false, false);
      }
    }
  }

#pragma unroll
  for (int t = 0; t < TT; ++t) {
    const int n0 = (wave + t * NWAVE) * 16;
#pragma unroll
    for (int rt = 0; rt < RT; ++rt) {
      if constexpr (RELU_TO_LDS) {
#pragma unroll
        for (int r = 0; r < 8; ++r) {
          float v = acc[rt][t][r];
          v = v > 0.0f ? v : 0.0f;
          store_act_frag(Dh + rt * (KTH * FRAG), Dl + rt * (KTH * FRAG),
                         mrow + r, n0 + mn, v);
        }
      } else {
#pragma unroll
        for (int r = 0; r < 8; ++r) {
          int rowi = r0 + rt * 16 + mrow + r;
          if (rowi < Nrows)
            outp[(size_t)rowi * HID + n0 + mn] = acc[rt][t][r];
        }
      }
    }
  }
}

__global__ __launch_bounds__(BLK) void gcn_mlp_kernel(
    const float* __restrict__ x,
    const unsigned short* __restrict__ W0h, const unsigned short* __restrict__ W0l,
    const unsigned short* __restrict__ W1h, const unsigned short* __restrict__ W1l,
    const unsigned short* __restrict__ W2h, const unsigned short* __restrict__ W2l,
    float* __restrict__ out, int Nrows) {
  __shared__ __align__(32) unsigned short aXh[RT * KTX * FRAG], aXl[RT * KTX * FRAG];
  __shared__ __align__(32) unsigned short aH0h[RT * KTH * FRAG], aH0l[RT * KTH * FRAG];
  __shared__ __align__(32) unsigned short aH1h[RT * KTH * FRAG], aH1l[RT * KTH * FRAG];

  const int tid = threadIdx.x;
  const int lane = tid & 31;
  const int wave = tid >> 5;
  const int r0 = blockIdx.x * (RT * 16);

  // Stage x tile [64 x 64] -> hi/lo A-fragment layout in LDS.
  // Tail block: clamp row index (duplicate last row; final stores are guarded).
  for (int i = tid; i < RT * 16 * EMB; i += BLK) {
    int m = i >> 6, k = i & 63;
    int rr = r0 + m;
    rr = rr < Nrows ? rr : Nrows - 1;
    float v = x[(size_t)rr * EMB + k];
    int rt = m >> 4;
    store_act_frag(aXh + rt * (KTX * FRAG), aXl + rt * (KTX * FRAG),
                   m & 15, k, v);
  }
  __syncthreads();

  // Layer 0: h0 = relu(x @ W0), K = 64
  gcn_layer<KTX, true>(aXh, aXl, W0h, W0l, aH0h, aH0l, nullptr, 0, 0, lane, wave);
  __syncthreads();
  // Layer 1: h1 = relu(h0 @ W1), K = 256
  gcn_layer<KTH, true>(aH0h, aH0l, W1h, W1l, aH1h, aH1l, nullptr, 0, 0, lane, wave);
  __syncthreads();
  // Layer 2: out = h1 @ W2, K = 256
  gcn_layer<KTH, false>(aH1h, aH1l, W2h, W2l, nullptr, nullptr, out, r0, Nrows,
                        lane, wave);
}

extern "C" void kernel_launch(void* const* d_in, const int* in_sizes, int n_in,
                              void* d_out, int out_size, void* d_ws, size_t ws_size,
                              hipStream_t stream) {
  const int*   row = (const int*)d_in[0];
  const int*   col = (const int*)d_in[1];
  const float* val = (const float*)d_in[2];
  const float* emb = (const float*)d_in[3];
  const float* W0  = (const float*)d_in[4];
  const float* W1  = (const float*)d_in[5];
  const float* W2  = (const float*)d_in[6];
  float* out = (float*)d_out;

  const int E = in_sizes[0];
  const int N = in_sizes[3] / EMB;   // 100000

  // d_ws layout: fp32 x accumulator, then bf16 hi/lo swizzled weights.
  float* x = (float*)d_ws;
  unsigned short* w0h = (unsigned short*)(x + (size_t)N * EMB);
  unsigned short* w0l = w0h + EMB * HID;
  unsigned short* w1h = w0l + EMB * HID;
  unsigned short* w1l = w1h + HID * HID;
  unsigned short* w2h = w1l + HID * HID;
  unsigned short* w2l = w2h + HID * HID;

  // Weight prep (deterministic, trivial cost ~0.6MB, amortized over 1563 blocks)
  gcn_wprep_kernel<<<(EMB * HID + 255) / 256, 256, 0, stream>>>(W0, w0h, w0l, EMB, HID);
  gcn_wprep_kernel<<<(HID * HID + 255) / 256, 256, 0, stream>>>(W1, w1h, w1l, HID, HID);
  gcn_wprep_kernel<<<(HID * HID + 255) / 256, 256, 0, stream>>>(W2, w2h, w2l, HID, HID);

  hipMemsetAsync(x, 0, (size_t)N * EMB * sizeof(float), stream);

  long spmm_threads = (long)E * 16;
  int spmm_blocks = (int)((spmm_threads + 255) / 256);
  gcn_spmm_kernel<<<spmm_blocks, 256, 0, stream>>>(row, col, val, emb, x, E);

  int mlp_blocks = (N + RT * 16 - 1) / (RT * 16);   // 1563
  gcn_mlp_kernel<<<mlp_blocks, BLK, 0, stream>>>(x, w0h, w0l, w1h, w1l,
                                                 w2h, w2l, out, N);
}
